// _MutualMultiheadNonlocalNd_63642825392178
// MI455X (gfx1250) — matrix operand order repaired and checked by reading, not compiled, not run on hardware
//
#include <hip/hip_runtime.h>
#include <hip/hip_fp16.h>

typedef __attribute__((ext_vector_type(16))) _Float16 v16h;
typedef __attribute__((ext_vector_type(8)))  float    v8f;
typedef __attribute__((ext_vector_type(4)))  int      v4i;

#define WMMA16(a, b, c) __builtin_amdgcn_wmma_f32_16x16x32_f16( \
    false, (a), false, (b), (short)0, (c), false, false)

#define AS1 __attribute__((address_space(1)))
#define AS3 __attribute__((address_space(3)))

// ---------------------------------------------------------------------------
// Problem constants
// ---------------------------------------------------------------------------
#define NB   8        // batch
#define CC   256      // channels
#define NN   2304     // spatial (48*48)
#define HD   128      // hidden per head
#define SCL  0.0625f  // 256^-0.5

static __device__ __forceinline__ v8f vzero() {
  v8f v;
#pragma unroll
  for (int i = 0; i < 8; ++i) v[i] = 0.0f;
  return v;
}

// A operand (16xK tile, f16), src row-major [row][k], pre-offset to (row0,k0).
// Layout: lanes 0-15 row=lane hold K {0..7,16..23}; lanes 16-31 hold {8..15,24..31}.
static __device__ __forceinline__ v16h load_A16(const _Float16* src, int ld) {
  const int lane = threadIdx.x & 31;
  const int row  = lane & 15;
  const int khi  = (lane >> 4) ? 8 : 0;
  const _Float16* p = src + (size_t)row * ld + khi;
  v16h a;
#pragma unroll
  for (int e = 0; e < 16; ++e) {
    const int k = e + ((e >= 8) ? 8 : 0);
    a[e] = p[k];
  }
  return a;
}

// B operand (Kx16 tile, f16) from TRANSPOSED storage src[col][k] (ld = k-stride
// of a column row). Layout: col = lane%16; lanes 0-15 K={0..7,16..23}, lanes 16-31 K={8..15,24..31}.
static __device__ __forceinline__ v16h load_BT16(const _Float16* src, int ld) {
  const int lane = threadIdx.x & 31;
  const int col  = lane & 15;
  const int khi  = (lane >> 4) ? 8 : 0;
  const _Float16* p = src + (size_t)col * ld + khi;
  v16h b;
#pragma unroll
  for (int e = 0; e < 16; ++e) b[e] = p[e + ((e >= 8) ? 8 : 0)];
  return b;
}

// 16-byte async global->LDS copy (gfx1250, tracked by ASYNCcnt).
static __device__ __forceinline__ void async_cp16B(const _Float16* g, _Float16* l) {
#if __has_builtin(__builtin_amdgcn_global_load_async_to_lds_b128)
  __builtin_amdgcn_global_load_async_to_lds_b128(
      (AS1 v4i*)(g), (AS3 v4i*)(l), 0, 0);
#else
  const unsigned loff = (unsigned)(unsigned long long)(AS3 void*)(l);
  asm volatile("global_load_async_to_lds_b128 %0, %1, off"
               :: "v"(loff), "v"(g) : "memory");
#endif
}

// ---------------------------------------------------------------------------
// Prep 1: convert embed weights to f16; build head-reduced, transposed out-w:
//   WeT[s][oc][c] = sum_{j<4} out_w_s[j*128 + c][oc]   (f16, [2][256][128])
// ---------------------------------------------------------------------------
__global__ void prep_weights_kernel(const float* __restrict__ ew0,
                                    const float* __restrict__ ew1,
                                    const float* __restrict__ ow0,
                                    const float* __restrict__ ow1,
                                    _Float16* __restrict__ Wemb,
                                    _Float16* __restrict__ WeT) {
  const int i = blockIdx.x * blockDim.x + threadIdx.x;
  if (i < 2 * 384 * CC) {
    const float* w = (i < 384 * CC) ? ew0 : ew1;
    Wemb[i] = (_Float16)w[i % (384 * CC)];
  }
  if (i < 2 * CC * HD) {
    const float* w = (i < CC * HD) ? ow0 : ow1;
    const int j  = i % (CC * HD);
    const int oc = j / HD;
    const int c  = j % HD;
    const float sum = w[(size_t)c * CC + oc] + w[(size_t)(c + 128) * CC + oc] +
                      w[(size_t)(c + 256) * CC + oc] + w[(size_t)(c + 384) * CC + oc];
    WeT[i] = (_Float16)sum;
  }
}

// ---------------------------------------------------------------------------
// Prep 2: transpose+convert inputs: Xpos[s][b][n][c] = (f16) x_s[b][c][n]
// ---------------------------------------------------------------------------
__global__ void prep_x_kernel(const float* __restrict__ x0,
                              const float* __restrict__ x1,
                              _Float16* __restrict__ Xpos) {
  const size_t i = (size_t)blockIdx.x * blockDim.x + threadIdx.x;
  const size_t total = (size_t)2 * NB * NN * CC;
  if (i >= total) return;
  const int c = (int)(i % CC);
  size_t t = i / CC;
  const int n = (int)(t % NN);
  t /= NN;
  const int b = (int)(t % NB);
  const int s = (int)(t / NB);
  const float* x = s ? x1 : x0;
  Xpos[i] = (_Float16)x[((size_t)b * CC + c) * NN + n];
}

// ---------------------------------------------------------------------------
// Embed GEMM: E[s][b][o][n] = Wemb[s] @ Xpos[s][b]^T + bias, scattered into
//   Qpos[s][b][n][c] (c<128), Kpos[s][b][n][c] (128..255), Vch[s][b][c][n] (256..383)
// One 16x16 tile per wave. Tiles per (s,b): (384/16)*(2304/16)=24*144=3456.
// ---------------------------------------------------------------------------
__global__ __launch_bounds__(128) void embed_gemm_kernel(
    const _Float16* __restrict__ Wemb, const _Float16* __restrict__ Xpos,
    const float* __restrict__ eb0, const float* __restrict__ eb1,
    _Float16* __restrict__ Qpos, _Float16* __restrict__ Kpos,
    _Float16* __restrict__ Vch) {
  const int wg   = blockIdx.x * 4 + (threadIdx.x >> 5);
  const int lane = threadIdx.x & 31;
  const int tile = wg % 3456;
  const int sb   = wg / 3456;  // 0..15
  const int s = sb >> 3;
  const int tm = tile % 24, tn = tile / 24;

  const _Float16* W = Wemb + (size_t)s * 384 * CC + (size_t)tm * 16 * CC;
  const _Float16* X = Xpos + ((size_t)sb * NN + (size_t)tn * 16) * CC;

  v8f acc = vzero();
#pragma unroll
  for (int kc = 0; kc < 8; ++kc) {
    const v16h a = load_A16(W + kc * 32, CC);
    const v16h x = load_BT16(X + kc * 32, CC);
    acc = WMMA16(a, x, acc);
  }

  const float* eb = s ? eb1 : eb0;
  const int col  = lane & 15;
  const int rofs = (lane >> 4) ? 8 : 0;
  const int n = tn * 16 + col;
  const size_t sbN = (size_t)sb * NN;
#pragma unroll
  for (int r = 0; r < 8; ++r) {
    const int o = tm * 16 + r + rofs;
    const float val = acc[r] + eb[o];
    if (o < 128) {
      Qpos[(sbN + n) * HD + o] = (_Float16)val;
    } else if (o < 256) {
      Kpos[(sbN + n) * HD + (o - 128)] = (_Float16)val;
    } else {
      Vch[((size_t)sb * HD + (o - 256)) * NN + n] = (_Float16)val;
    }
  }
}

// ---------------------------------------------------------------------------
// Flash cross-attention: output stream s uses q from stream (1-s), k/v from s.
// Each wave owns 16 query rows (8 waves = 128 rows / block). Keys streamed in
// 32-key blocks, double-buffered into LDS via async global->LDS copies so the
// tile is fetched ONCE per workgroup (not once per wave). ASYNCcnt + barriers
// provide the producer/consumer handoff; compute B-operands come from LDS.
// Result Hpos[s][b][n][c] (f16, position-major).
// ---------------------------------------------------------------------------
__global__ __launch_bounds__(256) void attn_kernel(
    const _Float16* __restrict__ Qpos, const _Float16* __restrict__ Kpos,
    const _Float16* __restrict__ Vch, _Float16* __restrict__ Hpos) {
  __shared__ _Float16 ldsK[2][32 * HD];   // K tile: [key][c], 8KB per buffer
  __shared__ _Float16 ldsV[2][HD * 32];   // V tile: [c][key], 8KB per buffer
  __shared__ _Float16 ldsP[8][16 * 32];   // per-wave P staging

  const int tid  = threadIdx.x;
  const int wave = tid >> 5;
  const int lane = tid & 31;
  const int col  = lane & 15;
  const int rofs = (lane >> 4) ? 8 : 0;
  const int s  = blockIdx.z;                 // output stream
  const int b  = blockIdx.y;
  const int qb = blockIdx.x * 128 + wave * 16;

  const _Float16* Q = Qpos + ((size_t)((1 - s) * NB + b) * NN + qb) * HD;
  const _Float16* K = Kpos + (size_t)(s * NB + b) * NN * HD;
  const _Float16* V = Vch + (size_t)(s * NB + b) * HD * NN;

  v16h Aq[4];
#pragma unroll
  for (int kc = 0; kc < 4; ++kc) Aq[kc] = load_A16(Q + kc * 32, HD);

  v8f O[8];
  float mi[8], li[8];
#pragma unroll
  for (int t = 0; t < 8; ++t) O[t] = vzero();
#pragma unroll
  for (int r = 0; r < 8; ++r) { mi[r] = -1e30f; li[r] = 0.0f; }

  _Float16* pw = &ldsP[wave][0];

  // ---- stage copy: 4 async b128 transfers per thread (2 for K, 2 for V) ----
  // K tile (32 rows x 128 halves) is one contiguous 8KB block in Kpos.
  // V tile is 128 rows of 64 bytes (row stride NN halves).
#define STAGE_KV(kbuf, vbuf, kbase)                                          \
  do {                                                                       \
    const _Float16* gk = K + (size_t)(kbase) * HD;                           \
    _Float16* lk = (kbuf);                                                   \
    _Float16* lv = (vbuf);                                                   \
    _Pragma("unroll")                                                        \
    for (int i_ = 0; i_ < 2; ++i_) {                                         \
      const int off_ = (tid + i_ * 256) * 8; /* halves */                    \
      async_cp16B(gk + off_, lk + off_);                                     \
    }                                                                        \
    _Pragma("unroll")                                                        \
    for (int i_ = 0; i_ < 2; ++i_) {                                         \
      const int j_   = tid + i_ * 256;  /* transfer 0..511 */                \
      const int row_ = j_ >> 2;         /* c row 0..127   */                 \
      const int off_ = (j_ & 3) * 8;    /* halves in row  */                 \
      async_cp16B(V + (size_t)row_ * NN + (kbase) + off_,                    \
                  lv + row_ * 32 + off_);                                    \
    }                                                                        \
  } while (0)

  int bf = 0;
  STAGE_KV(ldsK[0], ldsV[0], 0);

  for (int kb = 0; kb < NN; kb += 32) {
    const bool has_next = (kb + 32) < NN;
    if (has_next) {
      STAGE_KV(ldsK[bf ^ 1], ldsV[bf ^ 1], kb + 32);
      // 4 just-issued ops in flight; <=4 means the current tile has landed
      asm volatile("s_wait_asynccnt 0x4" ::: "memory");
    } else {
      asm volatile("s_wait_asynccnt 0x0" ::: "memory");
    }
    __syncthreads();  // publish this tile across all waves

    const _Float16* kt = ldsK[bf];
    const _Float16* vt = ldsV[bf];

    // logits: two 16x16 tiles over 32 keys, K=128 channels (B ops from LDS)
    v8f s0 = vzero(), s1 = vzero();
#pragma unroll
    for (int kc = 0; kc < 4; ++kc) {
      const v16h b0 = load_BT16(kt + kc * 32, HD);
      const v16h b1 = load_BT16(kt + 16 * HD + kc * 32, HD);
      s0 = WMMA16(Aq[kc], b0, s0);
      s1 = WMMA16(Aq[kc], b1, s1);
    }
    // streaming softmax (rows live across each 16-lane group)
#pragma unroll
    for (int r = 0; r < 8; ++r) {
      const float v0 = s0[r] * SCL;
      const float v1 = s1[r] * SCL;
      float rm = fmaxf(v0, v1);
#pragma unroll
      for (int msk = 1; msk < 16; msk <<= 1) rm = fmaxf(rm, __shfl_xor(rm, msk, 32));
      const float mn = fmaxf(mi[r], rm);
      const float al = __expf(mi[r] - mn);
      const float p0 = __expf(v0 - mn);
      const float p1 = __expf(v1 - mn);
      float rs = p0 + p1;
#pragma unroll
      for (int msk = 1; msk < 16; msk <<= 1) rs += __shfl_xor(rs, msk, 32);
      li[r] = li[r] * al + rs;
      mi[r] = mn;
#pragma unroll
      for (int t = 0; t < 8; ++t) O[t][r] *= al;
      const int rr = r + rofs;
      pw[rr * 32 + col]      = (_Float16)p0;
      pw[rr * 32 + col + 16] = (_Float16)p1;
    }
    // P was written cross-lane into this wave's private LDS tile: drain DScnt
    asm volatile("s_wait_dscnt 0x0" ::: "memory");
    const v16h Ap = load_A16(pw, 32);
#pragma unroll
    for (int t = 0; t < 8; ++t) {
      const v16h bv = load_BT16(vt + t * 16 * 32, 32);
      O[t] = WMMA16(Ap, bv, O[t]);
    }
    __syncthreads();  // all waves done with this buffer before it is refilled
    bf ^= 1;
  }
#undef STAGE_KV

  _Float16* Hout = Hpos + ((size_t)(s * NB + b) * NN + qb) * HD;
#pragma unroll
  for (int t = 0; t < 8; ++t) {
#pragma unroll
    for (int r = 0; r < 8; ++r) {
      const float h = O[t][r] / li[r];
      Hout[(size_t)(r + rofs) * HD + t * 16 + col] = (_Float16)h;
    }
  }
}

// ---------------------------------------------------------------------------
// Output GEMM + bias + residual: out[s][b][oc][m] =
//   sum_c Hpos[s][b][m][c] * WeT[s][oc][c] + out_b[s][oc] + x_s[b][oc][m]
// Tiles per (s,b): (2304/16)*(256/16) = 144*16 = 2304.
// ---------------------------------------------------------------------------
__global__ __launch_bounds__(128) void out_gemm_kernel(
    const _Float16* __restrict__ Hpos, const _Float16* __restrict__ WeT,
    const float* __restrict__ ob0, const float* __restrict__ ob1,
    const float* __restrict__ x0, const float* __restrict__ x1,
    float* __restrict__ out) {
  const int wg   = blockIdx.x * 4 + (threadIdx.x >> 5);
  const int lane = threadIdx.x & 31;
  const int tile = wg % 2304;
  const int sb   = wg / 2304;  // 0..15
  const int s = sb >> 3, b = sb & 7;
  const int tm = tile % 144, tn = tile / 144;

  const _Float16* H = Hpos + ((size_t)sb * NN + (size_t)tm * 16) * HD;
  const _Float16* W = WeT + (size_t)s * CC * HD + (size_t)tn * 16 * HD;

  v8f acc = vzero();
#pragma unroll
  for (int kc = 0; kc < 4; ++kc) {
    const v16h a = load_A16(H + kc * 32, HD);
    const v16h w = load_BT16(W + kc * 32, HD);
    acc = WMMA16(a, w, acc);
  }

  const float* ob = s ? ob1 : ob0;
  const float* x  = s ? x1 : x0;
  const int col  = lane & 15;
  const int rofs = (lane >> 4) ? 8 : 0;
  const int oc = tn * 16 + col;
  const float bias = ob[oc];
#pragma unroll
  for (int r = 0; r < 8; ++r) {
    const int m = tm * 16 + r + rofs;
    const float v = acc[r] + bias + x[((size_t)b * CC + oc) * NN + m];
    out[((size_t)sb * CC + oc) * NN + m] = v;
  }
}

// ---------------------------------------------------------------------------
// Host-side launch
// ---------------------------------------------------------------------------
extern "C" void kernel_launch(void* const* d_in, const int* in_sizes, int n_in,
                              void* d_out, int out_size, void* d_ws, size_t ws_size,
                              hipStream_t stream) {
  (void)in_sizes; (void)n_in; (void)out_size; (void)ws_size;

  const float* x0  = (const float*)d_in[0];
  const float* x1  = (const float*)d_in[1];
  const float* ew0 = (const float*)d_in[2];
  const float* eb0 = (const float*)d_in[3];
  const float* ew1 = (const float*)d_in[4];
  const float* eb1 = (const float*)d_in[5];
  const float* ow0 = (const float*)d_in[6];
  const float* ob0 = (const float*)d_in[7];
  const float* ow1 = (const float*)d_in[8];
  const float* ob1 = (const float*)d_in[9];
  float* out = (float*)d_out;

  // Workspace carve-up (all f16)
  _Float16* ws = (_Float16*)d_ws;
  const size_t SZ_WEMB = (size_t)2 * 384 * CC;       // 196608
  const size_t SZ_WET  = (size_t)2 * CC * HD;        // 65536
  const size_t SZ_XPOS = (size_t)2 * NB * NN * CC;   // 9437184
  const size_t SZ_QKVH = (size_t)2 * NB * NN * HD;   // 4718592 each
  _Float16* Wemb = ws;
  _Float16* WeT  = Wemb + SZ_WEMB;
  _Float16* Xpos = WeT + SZ_WET;
  _Float16* Qpos = Xpos + SZ_XPOS;
  _Float16* Kpos = Qpos + SZ_QKVH;
  _Float16* Vch  = Kpos + SZ_QKVH;
  _Float16* Hpos = Vch + SZ_QKVH;

  // 1) weight conversion / head-reduction
  {
    const int total = 2 * 384 * CC;
    prep_weights_kernel<<<(total + 255) / 256, 256, 0, stream>>>(
        ew0, ew1, ow0, ow1, Wemb, WeT);
  }
  // 2) input transpose + f32->f16
  {
    const size_t total = SZ_XPOS;
    prep_x_kernel<<<(unsigned)((total + 255) / 256), 256, 0, stream>>>(x0, x1, Xpos);
  }
  // 3) embed GEMM: 16 (s,b) * 3456 tiles, 4 waves/block
  embed_gemm_kernel<<<(16 * 3456) / 4, 128, 0, stream>>>(
      Wemb, Xpos, eb0, eb1, Qpos, Kpos, Vch);
  // 4) flash cross-attention: 18 query blocks x 8 batch x 2 streams
  attn_kernel<<<dim3(NN / 128, NB, 2), 256, 0, stream>>>(Qpos, Kpos, Vch, Hpos);
  // 5) output GEMM + bias + residual: 16 (s,b) * 2304 tiles, 4 waves/block
  out_gemm_kernel<<<(16 * 2304) / 4, 128, 0, stream>>>(
      Hpos, WeT, ob0, ob1, x0, x1, out);
}
